// GLA_50036368998927
// MI455X (gfx1250) — matrix operand order repaired and checked by reading, not compiled, not run on hardware
//
#include <hip/hip_runtime.h>
#include <hip/hip_bf16.h>

#define B_    2
#define L_    1024
#define H_    32
#define KVH_  8
#define D_    64
#define HID_  2048
#define G_    4
#define LOW_  16
#define CHUNK 64
#define NSPLIT 4              // split d_v into 4 groups of 16 -> 256 sequential chains
#define LDA   72              // padded bf16 LDS stride for 64-wide tiles
#define LDV   24              // padded bf16 LDS stride for 16-wide tiles
#define LDF   68              // padded f32 LDS stride for TDM-staged 64-wide tiles

typedef __attribute__((ext_vector_type(16))) __bf16 v16bf;
typedef __attribute__((ext_vector_type(8)))  float  v8f;
typedef __attribute__((ext_vector_type(4)))  unsigned int u32x4;
typedef __attribute__((ext_vector_type(8)))  unsigned int u32x8;

#if defined(__has_builtin)
#  if __has_builtin(__builtin_amdgcn_s_wait_tensorcnt)
#    define WAIT_TENSORCNT0() __builtin_amdgcn_s_wait_tensorcnt(0)
#  endif
#endif
#ifndef WAIT_TENSORCNT0
#  define WAIT_TENSORCNT0() asm volatile("s_wait_tensorcnt 0x0" ::: "memory")
#endif

// ---------------- Tensor Data Mover: 2D tile load, 2 descriptor groups ----------------
// Inline asm (portable across ROCm7.2 / amdgpu-toolchain builtin arity differences).
__device__ __forceinline__ void tdm_load_2d(u32x4 g0, u32x8 g1) {
  asm volatile("tensor_load_to_lds %0, %1" :: "s"(g0), "s"(g1) : "memory");
}

// D# group 0: count=1, lds_addr, 57-bit global byte address, type=2 ("image")
__device__ __forceinline__ u32x4 make_g0(unsigned lds_addr, unsigned long long gaddr) {
  u32x4 g;
  g[0] = 1u;                                   // count=1 valid descriptor
  g[1] = lds_addr;                             // LDS byte address
  g[2] = (unsigned)gaddr;                      // global_addr[31:0]
  g[3] = ((unsigned)(gaddr >> 32) & 0x01FFFFFFu) | (2u << 30);  // addr[56:32] | type=2
  return g;
}

// D# group 1 for a 64x64 f32 tile out of a 2048x2048 row-major f32 tensor,
// LDS padding: 4 DWORDs after every 64 DWORDs (row) -> ld = 68 f32.
__device__ __forceinline__ u32x8 make_g1_2048x2048_t64() {
  const unsigned TD0 = 2048, TD1 = 2048, TILE0 = 64, TILE1 = 64;
  const unsigned long long STR0 = 2048;
  u32x8 g;
  g[0] = (2u << 16)      // data_size = 4 bytes
       | (1u << 20)      // pad_enable
       | (5u << 22)      // pad_interval: 2^(5+1) = 64 DWORDs
       | (3u << 25);     // pad_amount : 3+1 = 4 DWORDs
  g[1] = (TD0 & 0xFFFFu) << 16;                 // tensor_dim0[15:0] @ bit48
  g[2] = (TD0 >> 16) | ((TD1 & 0xFFFFu) << 16); // tensor_dim0 hi | tensor_dim1 lo
  g[3] = (TD1 >> 16) | (TILE0 << 16);           // tensor_dim1 hi | tile_dim0
  g[4] = TILE1;                                 // tile_dim1 | tile_dim2(=0)
  g[5] = (unsigned)STR0;                        // tensor_dim0_stride[31:0]
  g[6] = (unsigned)(STR0 >> 32);                // stride hi16 | tensor_dim1_stride lo16(=0)
  g[7] = 0u;                                    // tensor_dim1_stride hi (unused, 2D)
  return g;
}

// ---------------- WMMA fragment builders (CDNA5 wave32 layouts) ----------------
// A 16x32 bf16: lane half=lane>>4, m=lane&15; element e -> K = half*16 + e
__device__ __forceinline__ v16bf frag_A(const __bf16* p, int ld, int m, int half, int k0) {
  v16bf a;
#pragma unroll
  for (int e = 0; e < 16; ++e) {
    int kk = (half << 4) + e;
    a[e] = p[m * ld + k0 + kk];
  }
  return a;
}
// A = X^T where X stored row-major [K][M]
__device__ __forceinline__ v16bf frag_AT(const __bf16* p, int ld, int m, int half, int k0) {
  v16bf a;
#pragma unroll
  for (int e = 0; e < 16; ++e) {
    int kk = (half << 4) + e;
    a[e] = p[(k0 + kk) * ld + m];
  }
  return a;
}
// B 32x16 bf16 from row-major [K][N]: lane n=lane&15, half=lane>>4; element e -> K = half*16 + e
__device__ __forceinline__ v16bf frag_B(const __bf16* p, int ld, int n, int half, int k0) {
  v16bf b;
#pragma unroll
  for (int e = 0; e < 16; ++e) {
    int kk = (half << 4) + e;
    b[e] = p[(k0 + kk) * ld + n];
  }
  return b;
}
// B = X^T where X stored row-major [N][K]
__device__ __forceinline__ v16bf frag_BT(const __bf16* p, int ld, int n, int half, int k0) {
  v16bf b;
#pragma unroll
  for (int e = 0; e < 16; ++e) {
    int kk = (half << 4) + e;
    b[e] = p[n * ld + k0 + kk];
  }
  return b;
}

#define WMMA_BF16(A, Bv, C) \
  __builtin_amdgcn_wmma_f32_16x16x32_bf16(false, (A), false, (Bv), (short)0, (C), false, false)

// ---------------- Kernel A: low-rank gate projection -> log_sigmoid/16 ----------------
__global__ __launch_bounds__(64)
void gk_proj_kernel(const float* __restrict__ hidden, const float* __restrict__ w1,
                    const float* __restrict__ w2, const float* __restrict__ b2,
                    float* __restrict__ gk_out) {
  __shared__ float red[64][LOW_];
  __shared__ float y[LOW_];
  int row = blockIdx.x;            // b*L + l
  int t = threadIdx.x;             // 0..63
  float acc[LOW_];
#pragma unroll
  for (int j = 0; j < LOW_; ++j) acc[j] = 0.f;
  const float* hrow = hidden + (size_t)row * HID_;
  for (int i = t; i < HID_; i += 64) {
    float hv = hrow[i];
    const float* w1r = w1 + (size_t)i * LOW_;
#pragma unroll
    for (int j = 0; j < LOW_; ++j) acc[j] += hv * w1r[j];
  }
#pragma unroll
  for (int j = 0; j < LOW_; ++j) red[t][j] = acc[j];
  __syncthreads();
  if (t < LOW_) {
    float s = 0.f;
    for (int i = 0; i < 64; ++i) s += red[i][t];
    y[t] = s;
  }
  __syncthreads();
  if (t < G_) {
    float z = b2[t];
#pragma unroll
    for (int j = 0; j < LOW_; ++j) z += y[j] * w2[j * G_ + t];
    float ls = fminf(z, 0.f) - log1pf(expf(-fabsf(z)));   // log_sigmoid
    gk_out[(size_t)row * G_ + t] = ls * (1.f / 16.f);
  }
}

// ---------------- Kernel B: chunked GLA scan (WMMA), d_v split 4-ways ----------------
__global__ __launch_bounds__(128)
void gla_scan_kernel(const float* __restrict__ q, const float* __restrict__ k,
                     const float* __restrict__ v, const float* __restrict__ gk,
                     const float* __restrict__ S0, float* __restrict__ o_att,
                     float* __restrict__ Sfinal) {
  __shared__ __bf16 sQ[CHUNK * LDA];   // Q~ [t][d]
  __shared__ __bf16 sK[CHUNK * LDA];   // K~ [s][d]
  __shared__ __bf16 sP[CHUNK * LDA];   // masked P [t][s]
  __shared__ __bf16 sV[CHUNK * LDV];   // V   [s][dv16]
  __shared__ __bf16 sS[D_    * LDV];   // S_prev [dk][dv16]
  __shared__ float  c_cum[CHUNK];

  int bh = blockIdx.x / NSPLIT;
  int vs = blockIdx.x % NSPLIT;        // which 16-col slice of d_v
  int b = bh / H_, h = bh % H_;
  int kvh = h >> 2;                    // h / G
  int gidx = h & 3;                    // h % G
  int tid = threadIdx.x, lane = tid & 31, wave = tid >> 5;
  int half = lane >> 4, l15 = lane & 15;
  const float SCALE = 0.125f;          // D^-0.5

  // running state slice (f32, WMMA C-layout): wave w owns dk rows [16w,16w+16)
  v8f S_acc = {0.f, 0.f, 0.f, 0.f, 0.f, 0.f, 0.f, 0.f};
  {
    const float* S0p = S0 + (size_t)bh * D_ * D_;
#pragma unroll
    for (int r = 0; r < 8; ++r) {
      int dk = wave * 16 + half * 8 + r;
      int dv = vs * 16 + l15;
      float val = S0p[dk * D_ + dv];
      S_acc[r] = val;
      sS[dk * LDV + l15] = (__bf16)val;
    }
  }
  __syncthreads();

  for (int ch = 0; ch < L_ / CHUNK; ++ch) {
    int l0 = ch * CHUNK;
    if (tid == 0) {                    // inclusive prefix of log-decay
      float cc = 0.f;
      for (int t = 0; t < CHUNK; ++t) {
        cc += gk[(size_t)(b * L_ + l0 + t) * G_ + gidx];
        c_cum[t] = cc;
      }
    }
    __syncthreads();

    // stage Q~, K~ (full 64x64) and V slice (64x16), bf16
    {
      int row = tid >> 1;
      int c0 = (tid & 1) * 32;
      float ct = c_cum[row];
      float qs = expf(ct) * SCALE;
      float ks = expf(-ct);
      size_t qoff  = ((size_t)(b * L_ + l0 + row) * H_   + h)   * D_;
      size_t kvoff = ((size_t)(b * L_ + l0 + row) * KVH_ + kvh) * D_;
#pragma unroll 8
      for (int j = 0; j < 32; ++j) {
        sQ[row * LDA + c0 + j] = (__bf16)(q[qoff + c0 + j] * qs);
        sK[row * LDA + c0 + j] = (__bf16)(k[kvoff + c0 + j] * ks);
      }
      if (tid < 64) {
#pragma unroll
        for (int j = 0; j < 16; ++j)
          sV[tid * LDV + j] = (__bf16)(v[((size_t)(b * L_ + l0 + tid) * KVH_ + kvh) * D_ + vs * 16 + j]);
      }
    }
    __syncthreads();

    // P = Q~ @ K~^T  (64x64), causal mask, store bf16
#pragma unroll
    for (int nt = 0; nt < 4; ++nt) {
      v8f acc = {0.f, 0.f, 0.f, 0.f, 0.f, 0.f, 0.f, 0.f};
#pragma unroll
      for (int ks2 = 0; ks2 < 2; ++ks2) {
        v16bf a  = frag_A (sQ, LDA, wave * 16 + l15, half, ks2 * 32);
        v16bf bb = frag_BT(sK, LDA, nt * 16 + l15,   half, ks2 * 32);
        acc = WMMA_BF16(a, bb, acc);
      }
#pragma unroll
      for (int r = 0; r < 8; ++r) {
        int t = wave * 16 + half * 8 + r;
        int s = nt * 16 + l15;
        sP[t * LDA + s] = (__bf16)((s <= t) ? acc[r] : 0.f);
      }
    }
    __syncthreads();

    // O = Q~ @ S_prev + P @ V   (64x16 slice)
    {
      v8f oacc = {0.f, 0.f, 0.f, 0.f, 0.f, 0.f, 0.f, 0.f};
#pragma unroll
      for (int ks2 = 0; ks2 < 2; ++ks2) {
        v16bf a  = frag_A(sQ, LDA, wave * 16 + l15, half, ks2 * 32);
        v16bf bb = frag_B(sS, LDV, l15, half, ks2 * 32);
        oacc = WMMA_BF16(a, bb, oacc);
      }
#pragma unroll
      for (int ks2 = 0; ks2 < 2; ++ks2) {
        v16bf a  = frag_A(sP, LDA, wave * 16 + l15, half, ks2 * 32);
        v16bf bb = frag_B(sV, LDV, l15, half, ks2 * 32);
        oacc = WMMA_BF16(a, bb, oacc);
      }
#pragma unroll
      for (int r = 0; r < 8; ++r) {
        int t = wave * 16 + half * 8 + r;
        o_att[((size_t)(b * L_ + l0 + t) * H_ + h) * D_ + vs * 16 + l15] = oacc[r];
      }
    }

    // dS = K~^T @ V ; S_new = e^{c_last} (S_prev + dS)
    {
#pragma unroll
      for (int ks2 = 0; ks2 < 2; ++ks2) {
        v16bf a  = frag_AT(sK, LDA, wave * 16 + l15, half, ks2 * 32);  // (K~^T)[dk][s]
        v16bf bb = frag_B (sV, LDV, l15, half, ks2 * 32);
        S_acc = WMMA_BF16(a, bb, S_acc);
      }
      float dec = expf(c_cum[CHUNK - 1]);
#pragma unroll
      for (int r = 0; r < 8; ++r) S_acc[r] *= dec;
    }
    __syncthreads();     // all waves done reading sS before overwrite
#pragma unroll
    for (int r = 0; r < 8; ++r)
      sS[(wave * 16 + half * 8 + r) * LDV + l15] = (__bf16)S_acc[r];
    __syncthreads();
  }

  // final state
  {
    float* Sf = Sfinal + (size_t)bh * D_ * D_;
#pragma unroll
    for (int r = 0; r < 8; ++r)
      Sf[(wave * 16 + half * 8 + r) * D_ + vs * 16 + l15] = S_acc[r];
  }
}

// ---- Kernel C: gate GEMM g = hidden @ g_proj_w, TDM-staged, bf16 WMMA, f32 accum ----
__global__ __launch_bounds__(128)
void gate_gemm_kernel(const float* __restrict__ hidden, const float* __restrict__ W,
                      float* __restrict__ gout) {
  __shared__ float  fA[64 * LDF];      // TDM staging (f32, padded ld=68)
  __shared__ float  fB[64 * LDF];
  __shared__ __bf16 sA[64 * LDA];      // compute tiles (bf16)
  __shared__ __bf16 sB[64 * LDA];

  int m0 = blockIdx.x * 64, n0 = blockIdx.y * 64;
  int tid = threadIdx.x, lane = tid & 31, wave = tid >> 5;
  int half = lane >> 4, l15 = lane & 15;

  unsigned ldsA = (unsigned)(size_t)(void*)fA;   // LDS byte offset (low 32 bits of generic addr)
  unsigned ldsB = (unsigned)(size_t)(void*)fB;
  unsigned long long baseA = (unsigned long long)(size_t)hidden;
  unsigned long long baseB = (unsigned long long)(size_t)W;
  u32x8 g1 = make_g1_2048x2048_t64();            // same tensor/tile geometry for A and B

  v8f acc[4];
#pragma unroll
  for (int nt = 0; nt < 4; ++nt)
#pragma unroll
    for (int r = 0; r < 8; ++r) acc[nt][r] = 0.f;

  // preload first K-tile via TDM (wave0 issues; EXEC-independent DMA, TENSORcnt-tracked)
  if (wave == 0) {
    tdm_load_2d(make_g0(ldsA, baseA + ((size_t)m0 * HID_ + 0) * 4ull), g1);
    tdm_load_2d(make_g0(ldsB, baseB + ((size_t)0 * (H_ * D_) + n0) * 4ull), g1);
  }
  WAIT_TENSORCNT0();
  __syncthreads();

  for (int kt = 0; kt < HID_ / 64; ++kt) {
    // convert staged f32 -> bf16 compute tiles
    {
      int row = tid >> 1, c0 = (tid & 1) * 32;
#pragma unroll 8
      for (int j = 0; j < 32; ++j) {
        sA[row * LDA + c0 + j] = (__bf16)fA[row * LDF + c0 + j];
        sB[row * LDA + c0 + j] = (__bf16)fB[row * LDF + c0 + j];
      }
    }
    __syncthreads();                      // staging consumed; compute tiles ready

    // kick off next tile's DMA, overlapping with the WMMA work below
    if (wave == 0 && kt + 1 < HID_ / 64) {
      int k0 = (kt + 1) * 64;
      tdm_load_2d(make_g0(ldsA, baseA + ((size_t)m0 * HID_ + k0) * 4ull), g1);
      tdm_load_2d(make_g0(ldsB, baseB + ((size_t)k0 * (H_ * D_) + n0) * 4ull), g1);
    }

#pragma unroll
    for (int nt = 0; nt < 4; ++nt)
#pragma unroll
      for (int ks2 = 0; ks2 < 2; ++ks2) {
        v16bf a  = frag_A(sA, LDA, wave * 16 + l15, half, ks2 * 32);
        v16bf bb = frag_B(sB, LDA, nt * 16 + l15,  half, ks2 * 32);
        acc[nt] = WMMA_BF16(a, bb, acc[nt]);
      }

    WAIT_TENSORCNT0();                    // next staging tile landed (no-op for waves 1-3)
    __syncthreads();                      // everyone done with compute tiles before re-convert
  }

#pragma unroll
  for (int nt = 0; nt < 4; ++nt)
#pragma unroll
    for (int r = 0; r < 8; ++r)
      gout[(size_t)(m0 + wave * 16 + half * 8 + r) * (H_ * D_) + n0 + nt * 16 + l15] = acc[nt][r];
}

// ---------------- Kernel D: LayerNorm(D=64) * SiLU(g), in place over d_out ----------------
__global__ __launch_bounds__(256)
void epilogue_kernel(const float* __restrict__ o_att, const float* __restrict__ gnw,
                     const float* __restrict__ gnb, float* __restrict__ out) {
  int row = blockIdx.x * 8 + (threadIdx.x >> 5);   // over B*L*H rows
  int lane = threadIdx.x & 31;
  const float* orow = o_att + (size_t)row * D_;
  float x0 = orow[lane], x1 = orow[lane + 32];
  float s = x0 + x1;
#pragma unroll
  for (int off = 16; off; off >>= 1) s += __shfl_xor(s, off, 32);
  float mu = s * (1.f / 64.f);
  float d0 = x0 - mu, d1 = x1 - mu;
  float vsum = d0 * d0 + d1 * d1;
#pragma unroll
  for (int off = 16; off; off >>= 1) vsum += __shfl_xor(vsum, off, 32);
  float inv = rsqrtf(vsum * (1.f / 64.f) + 1e-5f);
  float y0 = d0 * inv * gnw[lane] + gnb[lane];
  float y1 = d1 * inv * gnw[lane + 32] + gnb[lane + 32];
  float* op = out + (size_t)row * D_;
  float g0 = op[lane], g1 = op[lane + 32];
  op[lane]      = y0 * (g0 / (1.f + expf(-g0)));
  op[lane + 32] = y1 * (g1 / (1.f + expf(-g1)));
}

// ---------------- launch ----------------
extern "C" void kernel_launch(void* const* d_in, const int* in_sizes, int n_in,
                              void* d_out, int out_size, void* d_ws, size_t ws_size,
                              hipStream_t stream) {
  (void)in_sizes; (void)n_in; (void)out_size; (void)ws_size;
  const float* q      = (const float*)d_in[0];
  const float* k      = (const float*)d_in[1];
  const float* v      = (const float*)d_in[2];
  const float* hidden = (const float*)d_in[3];
  const float* S0     = (const float*)d_in[4];
  const float* g_proj = (const float*)d_in[5];
  const float* gk_w1  = (const float*)d_in[6];
  const float* gk_w2  = (const float*)d_in[7];
  const float* gk_b2  = (const float*)d_in[8];
  const float* gnw    = (const float*)d_in[9];
  const float* gnb    = (const float*)d_in[10];

  float* out    = (float*)d_out;
  float* Sfinal = out + (size_t)B_ * L_ * H_ * D_;
  float* gk_ws  = (float*)d_ws;                                    // B*L*G floats
  float* o_att  = (float*)((char*)d_ws + (size_t)B_ * L_ * G_ * sizeof(float));  // B*L*H*D floats

  gk_proj_kernel<<<B_ * L_, 64, 0, stream>>>(hidden, gk_w1, gk_w2, gk_b2, gk_ws);
  gla_scan_kernel<<<B_ * H_ * NSPLIT, 128, 0, stream>>>(q, k, v, gk_ws, S0, o_att, Sfinal);
  gate_gemm_kernel<<<dim3((B_ * L_) / 64, (H_ * D_) / 64), 128, 0, stream>>>(hidden, g_proj, out);
  epilogue_kernel<<<(B_ * L_ * H_) / 8, 256, 0, stream>>>(o_att, gnw, gnb, out);
}
